// SparseMoE_88055419502702
// MI455X (gfx1250) — matrix operand order repaired and checked by reading, not compiled, not run on hardware
//
#include <hip/hip_runtime.h>
#include <hip/hip_bf16.h>
#include <math.h>

// ---------------------------------------------------------------------------
// Sparse MoE for MI455X (gfx1250, wave32, WMMA, async-to-LDS).
// Compute-bound (~1400 FLOP/byte) -> v_wmma_f32_16x16x32_f16 with f32 accum.
// Top-2 sparsity exploited (4x less GEMM work than dense reference einsum).
// Weights/activations pre-converted once to f16 (pre-transposed [n][k]) so
// every GEMM tile load is a verbatim 16B copy done with
// GLOBAL_LOAD_ASYNC_TO_LDS_B128 + s_wait_asynccnt, double-buffered in LDS.
// Workspace requirement: ~60 MB.
// ---------------------------------------------------------------------------

typedef __attribute__((ext_vector_type(16))) _Float16 v16h;
typedef __attribute__((ext_vector_type(8)))  _Float16 v8h;
typedef __attribute__((ext_vector_type(8)))  float    v8f;

#define B_ROWS   4096
#define NEXP     8
#define DIN      1024
#define HID      1024
#define DOUT     512
#define TILE_M   128
#define MAXSLOT  4096                 // per-expert worst case
#define LDS_STRIDE 40                 // 32 + 8 halves pad (80B, 16B aligned)

// workspace layout (byte offsets)
#define WS_COUNTS   0u                                   // int[8]
#define WS_OFFSETS  64u                                  // int[16] padded prefix
#define WS_IMP      192u                                 // float[8]
#define WS_LOAD     256u                                 // float[8]
#define WS_ROWS     1024u                                // int[8*4096]
#define WS_GATES    (WS_ROWS + 4u*NEXP*MAXSLOT)          // float[8*4096]
#define WS_COMBINED (1u<<20)                             // float[4096*512] 8MB
#define WS_XH       (9u<<20)                             // f16[4096*1024] 8MB
#define WS_W1T      (17u<<20)                            // f16[8][1024n][1024k] 16MB
#define WS_W2T      (33u<<20)                            // f16[8][512n][1024k] 8MB
#define WS_H        (41u<<20)                            // f16[9216*1024] ~19MB

#define ASYNC_B128(ldsoff, goff, sbase)                                        \
  asm volatile("global_load_async_to_lds_b128 %0, %1, %2"                      \
               :: "v"(ldsoff), "v"(goff), "s"(sbase) : "memory")
#define WAIT_ASYNC_4() asm volatile("s_wait_asynccnt 0x4" ::: "memory")
#define WAIT_ASYNC_0() asm volatile("s_wait_asynccnt 0x0" ::: "memory")

// ---------------------------------------------------------------------------
__global__ void moe_init(char* ws) {
  int idx = blockIdx.x * blockDim.x + threadIdx.x;
  float* comb = (float*)(ws + WS_COMBINED);
  if (idx < B_ROWS * DOUT) comb[idx] = 0.f;
  if (idx < NEXP * MAXSLOT) {
    ((int*)(ws + WS_ROWS))[idx] = 0;
    ((float*)(ws + WS_GATES))[idx] = 0.f;
  }
  if (idx < NEXP) {
    ((int*)(ws + WS_COUNTS))[idx] = 0;
    ((float*)(ws + WS_IMP))[idx] = 0.f;
    ((float*)(ws + WS_LOAD))[idx] = 0.f;
  }
  if (idx < 16) ((int*)(ws + WS_OFFSETS))[idx] = 0;
}

// ---------------------------------------------------------------------------
// X (f32, row-major) -> Xh (f16, row-major), vectorized.
__global__ void moe_cvt_x(const float* __restrict__ x, char* ws) {
  _Float16* xh = (_Float16*)(ws + WS_XH);
  const int i = (blockIdx.x * blockDim.x + threadIdx.x) * 4;   // grid is exact
  const float4 v = *(const float4*)(x + i);
  _Float16* d = xh + i;
  d[0] = (_Float16)v.x; d[1] = (_Float16)v.y;
  d[2] = (_Float16)v.z; d[3] = (_Float16)v.w;
}

// ---------------------------------------------------------------------------
// Per-expert transpose + f32->f16: src [e][K][N] -> dst [e][N][K].
// 64x64 tiles staged in LDS; both global phases coalesced.
__global__ __launch_bounds__(256) void moe_transpose_cvt(
    const float* __restrict__ src, _Float16* __restrict__ dst, int K, int N) {
  const int e = blockIdx.z;
  const int kt = blockIdx.y * 64, nt = blockIdx.x * 64;
  src += (size_t)e * K * N;
  dst += (size_t)e * K * N;
  __shared__ _Float16 T[64 * 72];
  const int tid = threadIdx.x;
  const int r = tid >> 2, c = (tid & 3) * 16;
  {
    const float* s = src + (size_t)(kt + r) * N + nt + c;
#pragma unroll
    for (int j = 0; j < 16; j += 4) {
      const float4 v = *(const float4*)(s + j);
      T[r * 72 + c + j + 0] = (_Float16)v.x;
      T[r * 72 + c + j + 1] = (_Float16)v.y;
      T[r * 72 + c + j + 2] = (_Float16)v.z;
      T[r * 72 + c + j + 3] = (_Float16)v.w;
    }
  }
  __syncthreads();
  {
    _Float16* d = dst + (size_t)(nt + r) * K + kt + c;
#pragma unroll
    for (int j = 0; j < 16; ++j) d[j] = T[(c + j) * 72 + r];
  }
}

// ---------------------------------------------------------------------------
// One wave32 per row: noisy top-k gating + dispatch-list construction.
__global__ __launch_bounds__(256) void moe_gating(
    const float* __restrict__ x, const float* __restrict__ gate_scores,
    const float* __restrict__ noise, const float* __restrict__ w_noise,
    char* ws)
{
  const int wave = threadIdx.x >> 5, lane = threadIdx.x & 31;
  const int b = blockIdx.x * 8 + wave;            // grid is exact (512 blocks)
  const float* xrow = x + (size_t)b * DIN;

  float acc[NEXP];
#pragma unroll
  for (int e = 0; e < NEXP; ++e) acc[e] = 0.f;
  for (int i = lane; i < DIN; i += 32) {
    const float xv = xrow[i];
    const float* wrow = w_noise + (size_t)i * NEXP;
#pragma unroll
    for (int e = 0; e < NEXP; ++e) acc[e] = fmaf(xv, wrow[e], acc[e]);
  }
#pragma unroll
  for (int e = 0; e < NEXP; ++e)
#pragma unroll
    for (int off = 16; off > 0; off >>= 1)
      acc[e] += __shfl_xor(acc[e], off, 32);

  if (lane != 0) return;

  int*   counts = (int*)(ws + WS_COUNTS);
  float* imp    = (float*)(ws + WS_IMP);
  float* ldv    = (float*)(ws + WS_LOAD);
  int*   rows   = (int*)(ws + WS_ROWS);
  float* gv     = (float*)(ws + WS_GATES);

  const float* gsr = gate_scores + (size_t)b * NEXP;
  const float* nzr = noise + (size_t)b * NEXP;
  float sd[NEXP], nl[NEXP];
#pragma unroll
  for (int e = 0; e < NEXP; ++e) {
    const float s = acc[e];
    const float sp = (s > 20.f) ? s : log1pf(__expf(s));   // softplus
    sd[e] = sp + 0.01f;                                    // + NOISE_EPS
    nl[e] = gsr[e] + nzr[e] * sd[e];
  }
  // top-3 (first occurrence on ties, like lax.top_k)
  int idxs[3]; float vals[3]; unsigned used = 0;
  for (int k = 0; k < 3; ++k) {
    float best = -INFINITY; int bi = 0;
    for (int e = 0; e < NEXP; ++e)
      if (!((used >> e) & 1u) && nl[e] > best) { best = nl[e]; bi = e; }
    used |= 1u << bi; idxs[k] = bi; vals[k] = best;
  }
  // softmax over top-2
  const float mx = fmaxf(vals[0], vals[1]);
  const float e0 = __expf(vals[0] - mx), e1 = __expf(vals[1] - mx);
  const float inv = 1.f / (e0 + e1);
  const float g0 = e0 * inv, g1 = e1 * inv;
  unsafeAtomicAdd(&imp[idxs[0]], g0);
  unsafeAtomicAdd(&imp[idxs[1]], g1);
  const int s0 = atomicAdd(&counts[idxs[0]], 1);
  rows[idxs[0] * MAXSLOT + s0] = b; gv[idxs[0] * MAXSLOT + s0] = g0;
  const int s1 = atomicAdd(&counts[idxs[1]], 1);
  rows[idxs[1] * MAXSLOT + s1] = b; gv[idxs[1] * MAXSLOT + s1] = g1;
  // load via prob-in-top-k (normal CDF on clean gate scores)
  const float thr_in = vals[2], thr_out = vals[1];
#pragma unroll
  for (int e = 0; e < NEXP; ++e) {
    const float thr = (nl[e] > thr_in) ? thr_in : thr_out;
    const float z = (gsr[e] - thr) / sd[e];
    const float p = 0.5f * (1.f + erff(z * 0.70710678118654752f));
    unsafeAtomicAdd(&ldv[e], p);
  }
}

// ---------------------------------------------------------------------------
// Tile-padded prefix offsets for H layout + cv^2 load-balance loss.
__global__ void moe_scan_loss(char* ws, float* loss_out) {
  if (threadIdx.x != 0 || blockIdx.x != 0) return;
  const int* counts = (const int*)(ws + WS_COUNTS);
  int* offs = (int*)(ws + WS_OFFSETS);
  int a = 0;
  for (int e = 0; e < NEXP; ++e) {
    offs[e] = a;
    a += ((counts[e] + TILE_M - 1) / TILE_M) * TILE_M;
  }
  const float* imp = (const float*)(ws + WS_IMP);
  const float* ld  = (const float*)(ws + WS_LOAD);
  float loss = 0.f;
  for (int which = 0; which < 2; ++which) {
    const float* v = which ? ld : imp;
    float s = 0.f;
    for (int e = 0; e < NEXP; ++e) s += v[e];
    const float m = s * (1.f / NEXP);
    float var = 0.f;
    for (int e = 0; e < NEXP; ++e) { const float d = v[e] - m; var += d * d; }
    var *= (1.f / (NEXP - 1));                 // ddof=1
    loss += var / (m * m + 1e-10f);
  }
  loss_out[0] = loss;
}

// ---------------------------------------------------------------------------
// GEMM1: H[slot, 0:1024] = relu(Xg @ W1[e] + b1[e]) -> f16 in ws.
// Block = (expert e, 128-slot tile t); 8 waves, each owns a 32x64 C block.
// Tiles streamed with async-to-LDS, double-buffered.
__global__ __launch_bounds__(256) void moe_gemm1(char* ws,
                                                 const float* __restrict__ B1)
{
  const int e = blockIdx.y, t = blockIdx.x;
  const int cnt = ((const int*)(ws + WS_COUNTS))[e];
  if (t * TILE_M >= cnt) return;
  const int slotbase = ((const int*)(ws + WS_OFFSETS))[e] + t * TILE_M;
  const int* rows = (const int*)(ws + WS_ROWS) + e * MAXSLOT;
  _Float16* H = (_Float16*)(ws + WS_H);
  const unsigned long long abase = (unsigned long long)(uintptr_t)(ws + WS_XH);
  const unsigned long long bbase = (unsigned long long)(uintptr_t)(ws + WS_W1T)
                                 + (unsigned long long)e * (HID * DIN * 2);

  __shared__ __attribute__((aligned(16))) _Float16 Abuf[2][TILE_M * LDS_STRIDE];
  __shared__ __attribute__((aligned(16))) _Float16 Bbuf[2][TILE_M * LDS_STRIDE];
  __shared__ int ridx[TILE_M];

  const int tid = threadIdx.x;
  if (tid < TILE_M) ridx[tid] = rows[t * TILE_M + tid];
  __syncthreads();

  const int wave = tid >> 5, lane = tid & 31;
  const int wm = wave >> 1, wn = wave & 1;       // 4x2 wave grid
  const int lh = lane >> 4, lm = lane & 15;
  const int r0 = tid >> 2, cc = (tid & 3) * 8;   // loader role: row, 16B chunk
  const int rid0 = ridx[r0], rid1 = ridx[r0 + 64];

  auto issueA = [&](int p, int kk) {
    const unsigned l0 = (unsigned)(uintptr_t)&Abuf[p][r0 * LDS_STRIDE + cc];
    const unsigned g0 = (unsigned)(rid0 * DIN + kk + cc) * 2u;
    ASYNC_B128(l0, g0, abase);
    const unsigned l1 = (unsigned)(uintptr_t)&Abuf[p][(r0 + 64) * LDS_STRIDE + cc];
    const unsigned g1 = (unsigned)(rid1 * DIN + kk + cc) * 2u;
    ASYNC_B128(l1, g1, abase);
  };
  auto issueB = [&](int p, int kk, int nbase) {
    const unsigned l0 = (unsigned)(uintptr_t)&Bbuf[p][r0 * LDS_STRIDE + cc];
    const unsigned g0 = (unsigned)((nbase + r0) * DIN + kk + cc) * 2u;
    ASYNC_B128(l0, g0, bbase);
    const unsigned l1 = (unsigned)(uintptr_t)&Bbuf[p][(r0 + 64) * LDS_STRIDE + cc];
    const unsigned g1 = (unsigned)((nbase + r0 + 64) * DIN + kk + cc) * 2u;
    ASYNC_B128(l1, g1, bbase);
  };

  for (int nchunk = 0; nchunk < HID / 128; ++nchunk) {
    const int nbase = nchunk * 128;
    const v8f zero = {0.f,0.f,0.f,0.f,0.f,0.f,0.f,0.f};
    v8f c[2][4];
#pragma unroll
    for (int mi = 0; mi < 2; ++mi)
#pragma unroll
      for (int ni = 0; ni < 4; ++ni) c[mi][ni] = zero;

    int p = 0;
    issueA(0, 0); issueB(0, 0, nbase);
    for (int kk = 0; kk < DIN; kk += 32) {
      if (kk + 32 < DIN) {                        // prefetch next K tile
        issueA(p ^ 1, kk + 32);
        issueB(p ^ 1, kk + 32, nbase);
        WAIT_ASYNC_4();                           // current tile's 4 ops done
      } else {
        WAIT_ASYNC_0();
      }
      __syncthreads();

      v16h af[2], bf[4];
#pragma unroll
      for (int mi = 0; mi < 2; ++mi) {           // A frag: M=lm, K 8-granule/lh
        const int row = wm * 32 + mi * 16 + lm;
        const v8h lo = *(const v8h*)&Abuf[p][row * LDS_STRIDE + 8 * lh];
        const v8h hi = *(const v8h*)&Abuf[p][row * LDS_STRIDE + 16 + 8 * lh];
#pragma unroll
        for (int i = 0; i < 8; ++i) { af[mi][i] = lo[i]; af[mi][i+8] = hi[i]; }
      }
#pragma unroll
      for (int ni = 0; ni < 4; ++ni) {           // B frag: N=lm, K 8-granule/lh
        const int col = wn * 64 + ni * 16 + lm;
        const v8h lo = *(const v8h*)&Bbuf[p][col * LDS_STRIDE + 8 * lh];
        const v8h hi = *(const v8h*)&Bbuf[p][col * LDS_STRIDE + 16 + 8 * lh];
#pragma unroll
        for (int i = 0; i < 8; ++i) { bf[ni][i] = lo[i]; bf[ni][i+8] = hi[i]; }
      }
#pragma unroll
      for (int mi = 0; mi < 2; ++mi)
#pragma unroll
        for (int ni = 0; ni < 4; ++ni)
          c[mi][ni] = __builtin_amdgcn_wmma_f32_16x16x32_f16(
              false, af[mi], false, bf[ni], (short)0, c[mi][ni], false, false);
      __syncthreads();                            // buf p free for reuse
      p ^= 1;
    }
    // epilogue: bias + relu -> f16 H
#pragma unroll
    for (int mi = 0; mi < 2; ++mi)
#pragma unroll
      for (int ni = 0; ni < 4; ++ni) {
        const int hcol = nbase + wn * 64 + ni * 16 + lm;
        const float bias = B1[e * HID + hcol];
#pragma unroll
        for (int r = 0; r < 8; ++r) {
          const int M = wm * 32 + mi * 16 + r + 8 * lh;
          const float hv = fmaxf(c[mi][ni][r] + bias, 0.f);
          H[(size_t)(slotbase + M) * HID + hcol] = (_Float16)hv;
        }
      }
  }
}

// ---------------------------------------------------------------------------
// GEMM2: out = H @ W2[e] + b2[e]; combined[row] += gate * exp(out) (atomic).
__global__ __launch_bounds__(256) void moe_gemm2(char* ws,
                                                 const float* __restrict__ B2)
{
  const int e = blockIdx.y, t = blockIdx.x;
  const int cnt = ((const int*)(ws + WS_COUNTS))[e];
  if (t * TILE_M >= cnt) return;
  const int slotbase = ((const int*)(ws + WS_OFFSETS))[e] + t * TILE_M;
  const int* rows = (const int*)(ws + WS_ROWS) + e * MAXSLOT;
  const float* gts = (const float*)(ws + WS_GATES) + e * MAXSLOT;
  float* comb = (float*)(ws + WS_COMBINED);
  const unsigned long long abase = (unsigned long long)(uintptr_t)(ws + WS_H);
  const unsigned long long bbase = (unsigned long long)(uintptr_t)(ws + WS_W2T)
                                 + (unsigned long long)e * (DOUT * HID * 2);

  __shared__ __attribute__((aligned(16))) _Float16 Abuf[2][TILE_M * LDS_STRIDE];
  __shared__ __attribute__((aligned(16))) _Float16 Bbuf[2][TILE_M * LDS_STRIDE];
  __shared__ int ridx[TILE_M];
  __shared__ float gval[TILE_M];

  const int tid = threadIdx.x;
  if (tid < TILE_M) {
    ridx[tid] = rows[t * TILE_M + tid];
    gval[tid] = gts[t * TILE_M + tid];
  }
  __syncthreads();

  const int wave = tid >> 5, lane = tid & 31;
  const int wm = wave >> 1, wn = wave & 1;
  const int lh = lane >> 4, lm = lane & 15;
  const int r0 = tid >> 2, cc = (tid & 3) * 8;

  auto issueA = [&](int p, int kk) {
    const unsigned l0 = (unsigned)(uintptr_t)&Abuf[p][r0 * LDS_STRIDE + cc];
    const unsigned g0 = (unsigned)((slotbase + r0) * HID + kk + cc) * 2u;
    ASYNC_B128(l0, g0, abase);
    const unsigned l1 = (unsigned)(uintptr_t)&Abuf[p][(r0 + 64) * LDS_STRIDE + cc];
    const unsigned g1 = (unsigned)((slotbase + r0 + 64) * HID + kk + cc) * 2u;
    ASYNC_B128(l1, g1, abase);
  };
  auto issueB = [&](int p, int kk, int nbase) {
    const unsigned l0 = (unsigned)(uintptr_t)&Bbuf[p][r0 * LDS_STRIDE + cc];
    const unsigned g0 = (unsigned)((nbase + r0) * HID + kk + cc) * 2u;
    ASYNC_B128(l0, g0, bbase);
    const unsigned l1 = (unsigned)(uintptr_t)&Bbuf[p][(r0 + 64) * LDS_STRIDE + cc];
    const unsigned g1 = (unsigned)((nbase + r0 + 64) * HID + kk + cc) * 2u;
    ASYNC_B128(l1, g1, bbase);
  };

  for (int nchunk = 0; nchunk < DOUT / 128; ++nchunk) {
    const int nbase = nchunk * 128;
    const v8f zero = {0.f,0.f,0.f,0.f,0.f,0.f,0.f,0.f};
    v8f c[2][4];
#pragma unroll
    for (int mi = 0; mi < 2; ++mi)
#pragma unroll
      for (int ni = 0; ni < 4; ++ni) c[mi][ni] = zero;

    int p = 0;
    issueA(0, 0); issueB(0, 0, nbase);
    for (int kk = 0; kk < HID; kk += 32) {
      if (kk + 32 < HID) {
        issueA(p ^ 1, kk + 32);
        issueB(p ^ 1, kk + 32, nbase);
        WAIT_ASYNC_4();
      } else {
        WAIT_ASYNC_0();
      }
      __syncthreads();

      v16h af[2], bf[4];
#pragma unroll
      for (int mi = 0; mi < 2; ++mi) {
        const int row = wm * 32 + mi * 16 + lm;
        const v8h lo = *(const v8h*)&Abuf[p][row * LDS_STRIDE + 8 * lh];
        const v8h hi = *(const v8h*)&Abuf[p][row * LDS_STRIDE + 16 + 8 * lh];
#pragma unroll
        for (int i = 0; i < 8; ++i) { af[mi][i] = lo[i]; af[mi][i+8] = hi[i]; }
      }
#pragma unroll
      for (int ni = 0; ni < 4; ++ni) {
        const int col = wn * 64 + ni * 16 + lm;
        const v8h lo = *(const v8h*)&Bbuf[p][col * LDS_STRIDE + 8 * lh];
        const v8h hi = *(const v8h*)&Bbuf[p][col * LDS_STRIDE + 16 + 8 * lh];
#pragma unroll
        for (int i = 0; i < 8; ++i) { bf[ni][i] = lo[i]; bf[ni][i+8] = hi[i]; }
      }
#pragma unroll
      for (int mi = 0; mi < 2; ++mi)
#pragma unroll
        for (int ni = 0; ni < 4; ++ni)
          c[mi][ni] = __builtin_amdgcn_wmma_f32_16x16x32_f16(
              false, af[mi], false, bf[ni], (short)0, c[mi][ni], false, false);
      __syncthreads();
      p ^= 1;
    }
    // epilogue: gate*exp(out+bias) accumulated into combined (valid slots only)
#pragma unroll
    for (int mi = 0; mi < 2; ++mi)
#pragma unroll
      for (int ni = 0; ni < 4; ++ni) {
        const int ocol = nbase + wn * 64 + ni * 16 + lm;
        const float bias = B2[e * DOUT + ocol];
#pragma unroll
        for (int r = 0; r < 8; ++r) {
          const int M = wm * 32 + mi * 16 + r + 8 * lh;
          if (t * TILE_M + M < cnt) {
            const float ov = c[mi][ni][r] + bias;
            const float val = gval[M] * __expf(ov);
            unsafeAtomicAdd(&comb[(size_t)ridx[M] * DOUT + ocol], val);
          }
        }
      }
  }
}

// ---------------------------------------------------------------------------
__global__ void moe_finalize(const char* ws, float* y) {
  const int idx = blockIdx.x * blockDim.x + threadIdx.x;
  if (idx >= B_ROWS * DOUT) return;
  float c = ((const float*)(ws + WS_COMBINED))[idx];
  if (c == 0.f) c = 2.2204460492503131e-16f;   // np.finfo(float64).eps
  y[idx] = logf(c);
}

// ---------------------------------------------------------------------------
extern "C" void kernel_launch(void* const* d_in, const int* in_sizes, int n_in,
                              void* d_out, int out_size, void* d_ws, size_t ws_size,
                              hipStream_t stream) {
  const float* x  = (const float*)d_in[0];   // [4096,16,64] -> flat [4096,1024]
  const float* gs = (const float*)d_in[1];   // [4096,8]
  const float* nz = (const float*)d_in[2];   // [4096,8]
  const float* wn = (const float*)d_in[3];   // [1024,8]
  const float* W1 = (const float*)d_in[4];   // [8,1024,1024]
  const float* b1 = (const float*)d_in[5];   // [8,1024]
  const float* W2 = (const float*)d_in[6];   // [8,1024,512]
  const float* b2 = (const float*)d_in[7];   // [8,512]
  float* out = (float*)d_out;                // y [4096,512] then loss scalar
  char* ws = (char*)d_ws;

  moe_init<<<(B_ROWS * DOUT + 255) / 256, 256, 0, stream>>>(ws);
  moe_cvt_x<<<(B_ROWS * DIN / 4) / 256, 256, 0, stream>>>(x, ws);
  moe_transpose_cvt<<<dim3(HID / 64, DIN / 64, NEXP), 256, 0, stream>>>(
      W1, (_Float16*)(ws + WS_W1T), DIN, HID);
  moe_transpose_cvt<<<dim3(DOUT / 64, HID / 64, NEXP), 256, 0, stream>>>(
      W2, (_Float16*)(ws + WS_W2T), HID, DOUT);
  moe_gating<<<B_ROWS / 8, 256, 0, stream>>>(x, gs, nz, wn, ws);
  moe_scan_loss<<<1, 32, 0, stream>>>(ws, out + (size_t)B_ROWS * DOUT);
  dim3 g(MAXSLOT / TILE_M, NEXP);            // (32, 8); blocks early-exit
  moe_gemm1<<<g, 256, 0, stream>>>(ws, b1);
  moe_gemm2<<<g, 256, 0, stream>>>(ws, b2);
  moe_finalize<<<(B_ROWS * DOUT + 255) / 256, 256, 0, stream>>>(ws, out);
}
